// CXINGeneral_1425929142863
// MI455X (gfx1250) — matrix operand order repaired and checked by reading, not compiled, not run on hardware
//
#include <hip/hip_runtime.h>
#include <hip/hip_bf16.h>

typedef __bf16 bf16_t;
typedef __attribute__((ext_vector_type(16))) __bf16 v16bf;
typedef __attribute__((ext_vector_type(8)))  __bf16 v8bf;
typedef __attribute__((ext_vector_type(2)))  __bf16 v2bf;
typedef __attribute__((ext_vector_type(8)))  float  v8f;
typedef __attribute__((ext_vector_type(2)))  float  v2f;

#define N_T    50000
#define N_S    100000
#define E_NNZ  400000
#define IN_CH  128
#define OUT_CH 256

// f32 -> bf16 pack; lowers to v_cvt_pk_bf16_f32 on gfx1250 (confirmed in disasm)
__device__ __forceinline__ v8bf cvt8(float4 a, float4 b) {
    v8f f = {a.x, a.y, a.z, a.w, b.x, b.y, b.z, b.w};
    return __builtin_convertvector(f, v8bf);
}
__device__ __forceinline__ v2bf cvt2(float a, float b) {
    v2f f = {a, b};
    return __builtin_convertvector(f, v2bf);
}

// C[M,N] = act( A[M,K] @ B[K,N] + bias ), A optionally split in K as A0(K0)|A1(K-K0).
// Block tile 128x64, BK=32, 256 threads = 8 waves (4 m x 2 n); each wave owns a 32x32
// output patch: 2 A-frags x 2 B-frags -> 4 v_wmma_f32_16x16x32_bf16 per K-step.
// OOB A rows are clamped to row M-1 (their products land only in never-stored C rows),
// so the steady-state loop has no bounds branches at all.
__global__ __launch_bounds__(256)
void gemm_wmma_bf16(const float* __restrict__ A0, const float* __restrict__ A1, int K0,
                    const float* __restrict__ B, const float* __restrict__ bias,
                    float* __restrict__ C, int M, int N, int K, int relu)
{
    __shared__ bf16_t lA[128][32];  // [m][k]
    __shared__ bf16_t lB[64][32];   // transposed: [n][k]

    const int tid     = threadIdx.x;
    const int lane    = tid & 31;
    const int wave    = tid >> 5;
    const int rowBase = blockIdx.x * 128;
    const int nBase   = blockIdx.y * 64;

    const int mrow = (wave & 3) * 32;     // wave's 32-row patch
    const int nc0  = (wave >> 2) * 32;    // wave's 32-col patch

    v8f acc00 = 0.0f, acc01 = 0.0f, acc10 = 0.0f, acc11 = 0.0f;

    // cooperative loader indices
    const int ar  = tid >> 1;             // 0..127 : A tile row
    const int ac0 = (tid & 1) * 16;       // A col start (16 floats)
    const int bkp = tid >> 4;             // 0..15 : B k-pair (k = 2*bkp, 2*bkp+1)
    const int bn0 = (tid & 15) * 4;       // B n-col start (4 floats)

    const int arowIdx = min(rowBase + ar, M - 1);   // clamp: branch-free staging
    const int nPhase  = A1 ? 2 : 1;

    for (int ph = 0; ph < nPhase; ++ph) {
        const float* Asrc = ph ? A1 : A0;
        const int    kb   = ph ? K0 : 0;
        const int    ke   = ph ? K  : K0;
        const int    lda  = ph ? (K - K0) : K0;
        const float* arow = Asrc + (size_t)arowIdx * lda - kb;   // indexed by global k

        for (int k0 = kb; k0 < ke; k0 += 32) {
            __syncthreads();
            // ---- stage A tile (128x32) as bf16 (two b128 stores, no branches)
            {
                const float4* p = (const float4*)(arow + k0 + ac0);
                float4 f0 = p[0], f1 = p[1], f2 = p[2], f3 = p[3];
                *(v8bf*)&lA[ar][ac0]     = cvt8(f0, f1);
                *(v8bf*)&lA[ar][ac0 + 8] = cvt8(f2, f3);
            }
            // ---- stage B tile (32x64) transposed into lB[n][k], k-pairs packed
            {
                const int k = k0 + 2 * bkp;
                const float4 f0 = *(const float4*)(B + (size_t)k       * N + nBase + bn0);
                const float4 f1 = *(const float4*)(B + (size_t)(k + 1) * N + nBase + bn0);
                *(v2bf*)&lB[bn0 + 0][2 * bkp] = cvt2(f0.x, f1.x);
                *(v2bf*)&lB[bn0 + 1][2 * bkp] = cvt2(f0.y, f1.y);
                *(v2bf*)&lB[bn0 + 2][2 * bkp] = cvt2(f0.z, f1.z);
                *(v2bf*)&lB[bn0 + 3][2 * bkp] = cvt2(f0.w, f1.w);
            }
            __syncthreads();

            // ---- fragments per CDNA5 wave32 WMMA layouts
            const int m  = lane & 15;     // A: row M ; B: col N (within 16)
            const int hf = lane >> 4;     // half select
            union { v16bf v; v8bf h[2]; } fa0, fa1, fb0, fb1;
            // A 16x32: a[0..7] -> K = hf*8 + i ; a[8..15] -> K = 16 + hf*8 + i
            fa0.h[0] = *(const v8bf*)&lA[mrow + m][hf * 8];
            fa0.h[1] = *(const v8bf*)&lA[mrow + m][16 + hf * 8];
            fa1.h[0] = *(const v8bf*)&lA[mrow + 16 + m][hf * 8];
            fa1.h[1] = *(const v8bf*)&lA[mrow + 16 + m][16 + hf * 8];
            // B 32x16: b[0..7] -> K = hf*8 + i ; b[8..15] -> K = 16 + hf*8 + i
            fb0.h[0] = *(const v8bf*)&lB[nc0 + m][hf * 8];
            fb0.h[1] = *(const v8bf*)&lB[nc0 + m][16 + hf * 8];
            fb1.h[0] = *(const v8bf*)&lB[nc0 + 16 + m][hf * 8];
            fb1.h[1] = *(const v8bf*)&lB[nc0 + 16 + m][16 + hf * 8];

            acc00 = __builtin_amdgcn_wmma_f32_16x16x32_bf16(false, fa0.v, false, fb0.v,
                                                            (short)0, acc00, false, false);
            acc01 = __builtin_amdgcn_wmma_f32_16x16x32_bf16(false, fa0.v, false, fb1.v,
                                                            (short)0, acc01, false, false);
            acc10 = __builtin_amdgcn_wmma_f32_16x16x32_bf16(false, fa1.v, false, fb0.v,
                                                            (short)0, acc10, false, false);
            acc11 = __builtin_amdgcn_wmma_f32_16x16x32_bf16(false, fa1.v, false, fb1.v,
                                                            (short)0, acc11, false, false);
        }
    }

    // ---- epilogue: C/D f32 layout: lane -> N = lane&15, VGPR r -> M = r + 8*(lane>>4)
    const int nloc = lane & 15;
    const int rofs = (lane >> 4) * 8;
    #pragma unroll
    for (int tr = 0; tr < 2; ++tr) {
        #pragma unroll
        for (int tc = 0; tc < 2; ++tc) {
            int gcol = nBase + nc0 + tc * 16 + nloc;
            float bv = bias ? bias[gcol] : 0.0f;
            #pragma unroll
            for (int r = 0; r < 8; ++r) {
                int gr = rowBase + mrow + tr * 16 + rofs + r;
                if (gr < M) {
                    float v = (tr ? (tc ? acc11[r] : acc10[r])
                                  : (tc ? acc01[r] : acc00[r])) + bv;
                    if (relu) v = fmaxf(v, 0.0f);
                    C[(size_t)gr * N + gcol] = v;
                }
            }
        }
    }
}

// agg initialization: agg = x_target  (epsilon = 0 residual)
__global__ void copy_f4(const float4* __restrict__ src, float4* __restrict__ dst, int n4) {
    int i = blockIdx.x * blockDim.x + threadIdx.x;
    if (i < n4) dst[i] = src[i];
}

// One wave per edge: agg[row,:] += val * support[col,:]  (256 ch = 8 f32/lane)
__global__ __launch_bounds__(256)
void spmm_edges(const int* __restrict__ rows, const int* __restrict__ cols,
                const float* __restrict__ vals, const float* __restrict__ sup,
                float* __restrict__ agg, int ne)
{
    int e = blockIdx.x * 8 + (threadIdx.x >> 5);
    if (e >= ne) return;
    int lane = threadIdx.x & 31;
    int r = rows[e], c = cols[e];
    float v = vals[e];
    const float4* s = (const float4*)(sup + (size_t)c * OUT_CH) + lane * 2;
    float* d = agg + (size_t)r * OUT_CH + lane * 8;
    float4 s0 = s[0], s1 = s[1];
    unsafeAtomicAdd(d + 0, v * s0.x); unsafeAtomicAdd(d + 1, v * s0.y);
    unsafeAtomicAdd(d + 2, v * s0.z); unsafeAtomicAdd(d + 3, v * s0.w);
    unsafeAtomicAdd(d + 4, v * s1.x); unsafeAtomicAdd(d + 5, v * s1.y);
    unsafeAtomicAdd(d + 6, v * s1.z); unsafeAtomicAdd(d + 7, v * s1.w);
}

extern "C" void kernel_launch(void* const* d_in, const int* in_sizes, int n_in,
                              void* d_out, int out_size, void* d_ws, size_t ws_size,
                              hipStream_t stream) {
    (void)in_sizes; (void)n_in; (void)out_size; (void)ws_size;
    const float* x_t  = (const float*)d_in[0];
    const float* xs0  = (const float*)d_in[1];
    const float* xs1  = (const float*)d_in[2];
    const float* v0   = (const float*)d_in[3];
    const float* v1   = (const float*)d_in[4];
    const int*   r0   = (const int*)d_in[5];
    const int*   c0   = (const int*)d_in[6];
    const int*   r1   = (const int*)d_in[7];
    const int*   c1   = (const int*)d_in[8];
    const float* W0   = (const float*)d_in[9];
    const float* W1   = (const float*)d_in[10];
    const float* mW0  = (const float*)d_in[11];
    const float* mb0  = (const float*)d_in[12];
    const float* mW1  = (const float*)d_in[13];
    const float* mb1  = (const float*)d_in[14];
    const float* Wm   = (const float*)d_in[15];
    const float* bm   = (const float*)d_in[16];
    float* out = (float*)d_out;

    char* ws = (char*)d_ws;
    const size_t S_SUP = (size_t)N_S * OUT_CH * sizeof(float);   // 102.4 MB
    const size_t S_H   = (size_t)N_T * OUT_CH * sizeof(float);   //  51.2 MB
    float* sup0 = (float*)(ws);
    float* sup1 = (float*)(ws + S_SUP);
    float* agg0 = (float*)(ws + 2 * S_SUP);
    float* agg1 = (float*)(ws + 2 * S_SUP + S_H);
    float* tmp0 = sup0;   // support regions reused as MLP ping-pong after SpMM
    float* tmp1 = sup1;

    dim3 blk(256);
    dim3 gS((N_S + 127) / 128, OUT_CH / 64);
    dim3 gT((N_T + 127) / 128, OUT_CH / 64);

    // support_b = x_src_b @ W_b
    gemm_wmma_bf16<<<gS, blk, 0, stream>>>(xs0, nullptr, IN_CH, W0, nullptr, sup0,
                                           N_S, OUT_CH, IN_CH, 0);
    gemm_wmma_bf16<<<gS, blk, 0, stream>>>(xs1, nullptr, IN_CH, W1, nullptr, sup1,
                                           N_S, OUT_CH, IN_CH, 0);

    // agg_b = x_target
    const int n4 = N_T * OUT_CH / 4;
    copy_f4<<<(n4 + 255) / 256, blk, 0, stream>>>((const float4*)x_t, (float4*)agg0, n4);
    copy_f4<<<(n4 + 255) / 256, blk, 0, stream>>>((const float4*)x_t, (float4*)agg1, n4);

    // agg_b += A_b @ support_b  (edge-parallel atomic scatter)
    spmm_edges<<<(E_NNZ + 7) / 8, blk, 0, stream>>>(r0, c0, v0, sup0, agg0, E_NNZ);
    spmm_edges<<<(E_NNZ + 7) / 8, blk, 0, stream>>>(r1, c1, v1, sup1, agg1, E_NNZ);

    // MLP branch 0: agg0 -> tmp0 -> agg0 -> tmp0   (Linear + bias + ReLU)
    gemm_wmma_bf16<<<gT, blk, 0, stream>>>(agg0, nullptr, OUT_CH, mW0 + 0 * OUT_CH * OUT_CH,
                                           mb0 + 0 * OUT_CH, tmp0, N_T, OUT_CH, OUT_CH, 1);
    gemm_wmma_bf16<<<gT, blk, 0, stream>>>(tmp0, nullptr, OUT_CH, mW0 + 1 * OUT_CH * OUT_CH,
                                           mb0 + 1 * OUT_CH, agg0, N_T, OUT_CH, OUT_CH, 1);
    gemm_wmma_bf16<<<gT, blk, 0, stream>>>(agg0, nullptr, OUT_CH, mW0 + 2 * OUT_CH * OUT_CH,
                                           mb0 + 2 * OUT_CH, tmp0, N_T, OUT_CH, OUT_CH, 1);
    // MLP branch 1
    gemm_wmma_bf16<<<gT, blk, 0, stream>>>(agg1, nullptr, OUT_CH, mW1 + 0 * OUT_CH * OUT_CH,
                                           mb1 + 0 * OUT_CH, tmp1, N_T, OUT_CH, OUT_CH, 1);
    gemm_wmma_bf16<<<gT, blk, 0, stream>>>(tmp1, nullptr, OUT_CH, mW1 + 1 * OUT_CH * OUT_CH,
                                           mb1 + 1 * OUT_CH, agg1, N_T, OUT_CH, OUT_CH, 1);
    gemm_wmma_bf16<<<gT, blk, 0, stream>>>(agg1, nullptr, OUT_CH, mW1 + 2 * OUT_CH * OUT_CH,
                                           mb1 + 2 * OUT_CH, tmp1, N_T, OUT_CH, OUT_CH, 1);

    // merge: out = [h0 | h1] @ Wm + bm  (single split-K GEMM, no materialized concat)
    gemm_wmma_bf16<<<gT, blk, 0, stream>>>(tmp0, tmp1, OUT_CH, Wm, bm, out,
                                           N_T, OUT_CH, 2 * OUT_CH, 0);
}
